// FTVSRNet_72773925863451
// MI455X (gfx1250) — matrix operand order repaired and checked by reading, not compiled, not run on hardware
//
#include <hip/hip_runtime.h>

// ---------- CDNA5 WMMA types ----------
typedef __attribute__((ext_vector_type(16))) __bf16 v16bf;
typedef __attribute__((ext_vector_type(8)))  float  v8f;

// ---------- problem constants ----------
static constexpr int NB = 2;          // batch
static constexpr int TT = 4;          // frames
static constexpr int CC = 64;         // channels
static constexpr int HH = 256, WW = 256;
static constexpr int SS = 4;          // stride
static constexpr int HS = 64, WS = 64;
static constexpr int FF = HS * WS;    // 4096 patches per batch
static constexpr int LL = CC * SS * SS; // 1024 patch length
static constexpr int C3 = 3 * CC;     // 192 conv input channels
static constexpr int KTOT = C3 * 9;   // 1728 GEMM K
static constexpr int PAD = 16;        // x halo (zeros) in the intermediate
static constexpr int WP  = WW + 2 * PAD;  // 288 padded width

// ---------- workspace layout (bytes) ----------
static constexpr size_t OFF_SRC   = 0;                              // int32 [NB][TT][FF]
static constexpr size_t OFF_CIDX  = OFF_SRC   + (size_t)NB*TT*FF*4; // int32 [NB][FF]
static constexpr size_t OFF_CSOFT = OFF_CIDX  + (size_t)NB*FF*4;    // f32   [NB][FF]
static constexpr size_t OFF_WB    = OFF_CSOFT + (size_t)NB*FF*4;    // bf16  [CC][KTOT]
static constexpr size_t OFF_CAT   = OFF_WB    + (size_t)CC*KTOT*2;  // bf16  [NB][HH][WP][C3]

__device__ inline unsigned short f2bf(float f) {
    unsigned u = __float_as_uint(f);
    u += 0x7FFFu + ((u >> 16) & 1u);   // round-to-nearest-even
    return (unsigned short)(u >> 16);
}

__device__ inline float wave_sum(float v) {
    v += __shfl_xor(v, 16, 32);
    v += __shfl_xor(v, 8, 32);
    v += __shfl_xor(v, 4, 32);
    v += __shfl_xor(v, 2, 32);
    v += __shfl_xor(v, 1, 32);
    return v;
}

// ============ K0: nearest-sample source index per (n,t,f) ============
__global__ void k_grid(const float* __restrict__ loc, int* __restrict__ src_idx) {
    int i = blockIdx.x * blockDim.x + threadIdx.x;
    if (i >= NB * TT * FF) return;
    int f = i % FF, nt = i / FF;
    int t = nt % TT, n = nt / TT;
    const float* base = loc + (size_t)n * (2 * TT) * FF;
    float lx = base[(size_t)(t * 2 + 0) * FF + f];
    float ly = base[(size_t)(t * 2 + 1) * FF + f];
    // grid_sample nearest, align_corners=True: pix = ((2*l/W - 1)+1)/2*(W-1) = l*(W-1)/W
    int ix = (int)rintf(lx * (float)(WS - 1) / (float)WS);
    int iy = (int)rintf(ly * (float)(HS - 1) / (float)HS);
    bool ok = (ix >= 0) && (ix < WS) && (iy >= 0) && (iy < HS);
    src_idx[i] = ok ? (iy * WS + ix) : -1;
}

// ============ K1: cosine corr over t, max/argmax; one wave per (n,f) ============
__global__ __launch_bounds__(256) void k_corr(const float* __restrict__ curr,
                                              const float* __restrict__ idxset,
                                              const int* __restrict__ src_idx,
                                              float* __restrict__ corr_soft,
                                              int* __restrict__ corr_index) {
    int tid  = threadIdx.x;
    int lane = tid & 31;
    int wid  = blockIdx.x * (blockDim.x >> 5) + (tid >> 5);
    if (wid >= NB * FF) return;
    int n = wid / FF, f = wid % FF;
    int fy = f / WS, fx = f % WS;

    float cf[32];
    float csq = 0.f;
#pragma unroll
    for (int j = 0; j < 32; j++) {
        int l = lane + 32 * j;                    // l = cc*16 + kh*4 + kw
        int cc = l >> 4, kh = (l >> 2) & 3, kw = l & 3;
        float v = curr[(((size_t)n * CC + cc) * HH + (fy * SS + kh)) * WW + (fx * SS + kw)];
        cf[j] = v;
        csq += v * v;
    }
    csq = wave_sum(csq);
    float ncf = fmaxf(sqrtf(csq), 1e-12f);

    float best = -3.0e38f;
    int bi = 0;
    for (int t = 0; t < TT; t++) {
        int src = src_idx[(size_t)(n * TT + t) * FF + f];   // uniform across wave
        float dot = 0.f, nsq = 0.f;
        if (src >= 0) {
            const float* bp = idxset + ((size_t)(n * TT + t) * LL) * FF + src;
#pragma unroll
            for (int j = 0; j < 32; j++) {
                int l = lane + 32 * j;
                float v = bp[(size_t)l * FF];
                dot += v * cf[j];
                nsq += v * v;
            }
        }
        dot = wave_sum(dot);
        nsq = wave_sum(nsq);
        float corr = dot / (ncf * fmaxf(sqrtf(nsq), 1e-12f));
        if (corr > best) { best = corr; bi = t; }  // first-max == jnp.argmax
    }
    if (lane == 0) {
        corr_soft[(size_t)n * FF + f]  = best;
        corr_index[(size_t)n * FF + f] = bi;
    }
}

// ============ KW: repack fusion_w [64][192][3][3] -> bf16 [co][tap*192+ci] ============
__global__ void k_wb(const float* __restrict__ w, unsigned short* __restrict__ wb) {
    int i = blockIdx.x * blockDim.x + threadIdx.x;
    if (i >= CC * KTOT) return;
    int co = i / KTOT, k = i % KTOT;
    int tap = k / C3, ci = k % C3;
    int kh = tap / 3, kw = tap % 3;
    wb[i] = f2bf(w[(((size_t)co * C3 + ci) * 3 + kh) * 3 + kw]);
}

// ============ K2: gather argmax-selected patches, fold -> bf16 channels-last,
//               x-padded cat [NB][HH][WP][C3] (halo columns zeroed) ============
__global__ void k_cat(const float* __restrict__ s1, const float* __restrict__ s2,
                      const float* __restrict__ s3, const int* __restrict__ src_idx,
                      const int* __restrict__ corr_index, unsigned short* __restrict__ cat) {
    int i = blockIdx.x * blockDim.x + threadIdx.x;  // one padded (n,y,xp) per thread
    if (i >= NB * HH * WP) return;
    int xp = i % WP, y = (i / WP) % HH, n = i / (WP * HH);
    int x = xp - PAD;
    unsigned short* o = cat + (size_t)i * C3;

    int src = -1, ti = 0;
    if (x >= 0 && x < WW) {
        int f = (y >> 2) * WS + (x >> 2);
        ti  = corr_index[(size_t)n * FF + f];
        src = src_idx[(size_t)(n * TT + ti) * FF + f];
    }
    if (src < 0) {                                   // halo or invalid sample -> zeros
        uint4 z; z.x = z.y = z.z = z.w = 0u;
#pragma unroll
        for (int q = 0; q < C3 / 8; q++) ((uint4*)o)[q] = z;
        return;
    }
    int rem = (y & 3) * 4 + (x & 3);                 // kh*4+kw part of l
    const float* sets[3] = { s1, s2, s3 };
#pragma unroll
    for (int k3 = 0; k3 < 3; k3++) {
        const float* sp = sets[k3] + ((size_t)(n * TT + ti) * LL + rem) * FF + src;
        union { unsigned short s[8]; uint4 q; } buf;
        for (int cc = 0; cc < CC; cc++) {
            buf.s[cc & 7] = f2bf(sp[(size_t)cc * 16 * FF]);   // l = cc*16 + rem
            if ((cc & 7) == 7)
                *(uint4*)(o + k3 * CC + (cc & ~7)) = buf.q;
        }
    }
}

// ============ K3: implicit-GEMM 3x3 conv (192->64) via bf16 WMMA + fused epilogue ======
// block = 256 thr (8 waves); tile = 64 co x 64 x; wave -> 16 co x 32 x (2 accums)
// padded input: no per-lane bounds checks -> EXEC stays all-ones through the K loop
__global__ __launch_bounds__(256) void k_conv(const unsigned short* __restrict__ wb,
                                              const unsigned short* __restrict__ cat,
                                              const float* __restrict__ bias,
                                              const float* __restrict__ corr_soft,
                                              const float* __restrict__ anchor,
                                              float* __restrict__ out) {
    int b  = blockIdx.x;                 // NB*HH*(WW/64) = 2048
    int xt = (b & 3) * 64;
    int y  = (b >> 2) & (HH - 1);
    int n  = b >> 10;
    int tid = threadIdx.x;
    int lane = tid & 31;
    int wv = tid >> 5;                   // 0..7
    int mw = wv & 3;                     // co tile (16 each)
    int nw = wv >> 2;                    // 0..1 -> x offset 32*nw
    int m16 = lane & 15;
    int g   = lane >> 4;

    v8f acc[2] = {};
    int co_a = mw * 16 + m16;            // row of A this lane loads

    union Frag { v16bf v; uint4 q[2]; };

    for (int tap = 0; tap < 9; tap++) {
        int kh = tap / 3, kw = tap % 3;
        int yy = y + kh - 1;
        if (yy < 0 || yy >= HH) continue;                 // SAME zero rows (uniform branch)
        const unsigned short* rowp = cat + (size_t)(n * HH + yy) * WP * C3;

        // lane base pointers for this tap; k-steps use constant immediate offsets
        const unsigned short* ap  = wb + (size_t)co_a * KTOT + tap * C3 + 8 * g;
        const unsigned short* bp0 = rowp
            + (size_t)(xt + nw * 32 + m16 + kw - 1 + PAD) * C3 + 8 * g;
        const unsigned short* bp1 = bp0 + 16 * C3;

        // pull next tap's row toward the WGP while this tap computes
        if (tap < 8) {
            int yn = y + (tap + 1) / 3 - 1;
            if (yn >= 0 && yn < HH) {
                const unsigned short* np = cat + (size_t)(n * HH + yn) * WP * C3
                    + (size_t)(xt + nw * 32 + m16 + PAD) * C3 + 16 * g;
                __builtin_prefetch(np, 0, 0);             // global_prefetch_b8
            }
        }

#pragma unroll
        for (int ks = 0; ks < 6; ks++) {
            // A fragment: 16-bit A layout -> lane(m,g): K = (e&7)+8g+16*(e>>3)
            Frag a;
            a.q[0] = *(const uint4*)(ap + ks * 32);       // e0..7  : K = kbase+8g+0..7
            a.q[1] = *(const uint4*)(ap + ks * 32 + 16);  // e8..15 : K = kbase+8g+16..23
            // B fragments: K = (e&7)+8g+16*(e>>3) (matches A), unconditional loads
            Frag b0;
            b0.q[0] = *(const uint4*)(bp0 + ks * 32);
            b0.q[1] = *(const uint4*)(bp0 + ks * 32 + 16);
            acc[0] = __builtin_amdgcn_wmma_f32_16x16x32_bf16(
                false, a.v, false, b0.v, (short)0, acc[0], false, false);
            Frag b1;
            b1.q[0] = *(const uint4*)(bp1 + ks * 32);
            b1.q[1] = *(const uint4*)(bp1 + ks * 32 + 16);
            acc[1] = __builtin_amdgcn_wmma_f32_16x16x32_bf16(
                false, a.v, false, b1.v, (short)0, acc[1], false, false);
        }
    }

    // epilogue: D layout -> VGPR r: M = r + 8g, N = lane&15
#pragma unroll
    for (int nt = 0; nt < 2; nt++) {
        int x = xt + nw * 32 + nt * 16 + m16;
        int f = (y >> 2) * WS + (x >> 2);
        float csf = corr_soft[(size_t)n * FF + f];
#pragma unroll
        for (int r = 0; r < 8; r++) {
            int co = mw * 16 + r + 8 * g;
            size_t oi = (((size_t)n * CC + co) * HH + y) * WW + x;
            out[oi] = (acc[nt][r] + bias[co]) * csf + anchor[oi];
        }
    }
}

// =====================================================================
extern "C" void kernel_launch(void* const* d_in, const int* in_sizes, int n_in,
                              void* d_out, int out_size, void* d_ws, size_t ws_size,
                              hipStream_t stream) {
    const float* curr   = (const float*)d_in[0];
    const float* idxset = (const float*)d_in[1];
    const float* anchor = (const float*)d_in[2];
    const float* s1     = (const float*)d_in[3];
    const float* s2     = (const float*)d_in[4];
    const float* s3     = (const float*)d_in[5];
    const float* loc    = (const float*)d_in[6];
    const float* fw     = (const float*)d_in[7];
    const float* fb     = (const float*)d_in[8];
    float* out = (float*)d_out;

    char* ws = (char*)d_ws;
    int*   src_idx = (int*)  (ws + OFF_SRC);
    int*   cidx    = (int*)  (ws + OFF_CIDX);
    float* csoft   = (float*)(ws + OFF_CSOFT);
    unsigned short* wb  = (unsigned short*)(ws + OFF_WB);
    unsigned short* cat = (unsigned short*)(ws + OFF_CAT);

    k_grid<<<(NB * TT * FF + 255) / 256, 256, 0, stream>>>(loc, src_idx);
    k_corr<<<(NB * FF) / 8, 256, 0, stream>>>(curr, idxset, src_idx, csoft, cidx);
    k_wb  <<<(CC * KTOT + 255) / 256, 256, 0, stream>>>(fw, wb);
    k_cat <<<(NB * HH * WP + 255) / 256, 256, 0, stream>>>(s1, s2, s3, src_idx, cidx, cat);
    k_conv<<<NB * HH * (WW / 64), 256, 0, stream>>>(wb, cat, fb, csoft, anchor, out);
}
